// Codebook_27118423506980
// MI455X (gfx1250) — matrix operand order repaired and checked by reading, not compiled, not run on hardware
//
#include <hip/hip_runtime.h>

typedef __attribute__((ext_vector_type(16))) _Float16 v16h;
typedef __attribute__((ext_vector_type(8)))  float    v8f;

#define K_NUM 8192
#define C_DIM 256
#define DHW   4096
#define B_SZ  8
#define OUT_ELEMS (B_SZ * C_DIM * DHW)   // 8388608

union Frag { v16h v; uint4 u4[2]; };

__device__ inline float blockReduceSum(float v) {
  #pragma unroll
  for (int off = 16; off > 0; off >>= 1) v += __shfl_down(v, off, 32);
  __shared__ float sh[8];
  int lane = threadIdx.x & 31, wv = threadIdx.x >> 5;
  if (lane == 0) sh[wv] = v;
  __syncthreads();
  v = (threadIdx.x < 8) ? sh[threadIdx.x] : 0.0f;
  if (wv == 0) {
    #pragma unroll
    for (int off = 4; off > 0; off >>= 1) v += __shfl_down(v, off, 32);
  }
  return v;  // valid in thread 0
}

// codebook: f32 -> f16 copy + per-row squared norm
__global__ __launch_bounds__(256) void k_cbconv(const float* __restrict__ cb,
                                                _Float16* __restrict__ ch,
                                                float* __restrict__ cn) {
  int k = blockIdx.x, c = threadIdx.x;
  float v = cb[k * C_DIM + c];
  ch[k * C_DIM + c] = (_Float16)v;
  float s = blockReduceSum(v * v);
  if (threadIdx.x == 0) cn[k] = s;
}

// z [B, C, DHW] f32 -> z_flat [N=B*DHW, C] f16, LDS-tiled transpose
__global__ __launch_bounds__(256) void k_zconv(const float* __restrict__ z,
                                               _Float16* __restrict__ zh) {
  __shared__ float tile[64][65];
  int b = blockIdx.z;
  int cBase = blockIdx.y * 64;
  int dBase = blockIdx.x * 64;
  int tx = threadIdx.x & 63;        // dhw within tile
  int ty = threadIdx.x >> 6;        // 0..3
  #pragma unroll
  for (int i = 0; i < 16; ++i) {
    int cl = ty + i * 4;
    tile[cl][tx] = z[((size_t)(b * C_DIM + cBase + cl)) * DHW + dBase + tx];
  }
  __syncthreads();
  int cu = threadIdx.x & 31;        // packed-pair index within 64 c's
  int r0 = threadIdx.x >> 5;        // 0..7
  unsigned* zp = (unsigned*)zh;
  #pragma unroll
  for (int j = 0; j < 8; ++j) {
    int dl = r0 + j * 8;
    int n = b * DHW + dBase + dl;
    union { _Float16 h[2]; unsigned u; } p;
    p.h[0] = (_Float16)tile[2 * cu][dl];
    p.h[1] = (_Float16)tile[2 * cu + 1][dl];
    zp[(size_t)n * 128 + (cBase >> 1) + cu] = p.u;
  }
}

// fused distance-GEMM + argmin.
// grid = 128 blocks x 8 waves, 2 M-tiles (32 rows) per wave:
// every B fragment read from LDS feeds two WMMAs.
__global__ __launch_bounds__(256) void k_argmin(const _Float16* __restrict__ zh,
                                                const _Float16* __restrict__ ch,
                                                const float* __restrict__ cn,
                                                int* __restrict__ idx) {
  __shared__ uint4 smem[2][16 * 33];          // 16 rows x 264 halves (padded)
  const uint4* zb4 = (const uint4*)zh;        // 32 uint4 per 256-half row
  const uint4* cb4 = (const uint4*)ch;
  int lane = threadIdx.x & 31, wv = threadIdx.x >> 5;
  int hl  = lane >> 4;                        // half of wave
  int l16 = lane & 15;
  int rowBase = blockIdx.x * 256 + wv * 32;

  // A fragments for K=0..255, two 16-row tiles (2 x 8 x 8 VGPRs)
  Frag a0[8], a1[8];
  int ar0 = rowBase + l16;
  int ar1 = rowBase + 16 + l16;
  #pragma unroll
  for (int kt = 0; kt < 8; ++kt) {
    a0[kt].u4[0] = zb4[(size_t)ar0 * 32 + kt * 4 + hl];       // K base + hl*8
    a0[kt].u4[1] = zb4[(size_t)ar0 * 32 + kt * 4 + 2 + hl];   // K base + 16 + hl*8
    a1[kt].u4[0] = zb4[(size_t)ar1 * 32 + kt * 4 + hl];
    a1[kt].u4[1] = zb4[(size_t)ar1 * 32 + kt * 4 + 2 + hl];
  }

  float bestD[2][8];
  int   bestI[2][8];
  #pragma unroll
  for (int t = 0; t < 2; ++t)
    #pragma unroll
    for (int v = 0; v < 8; ++v) { bestD[t][v] = 3.4e38f; bestI[t][v] = 0x7fffffff; }

  // preload codebook tile 0 into buffer 0 (512 uint4, 2 per thread)
  {
    int j0 = threadIdx.x * 2;
    #pragma unroll
    for (int e = 0; e < 2; ++e) {
      int j = j0 + e, r = j >> 5, col = j & 31;
      smem[0][r * 33 + col] = cb4[r * 32 + col];
    }
  }

  for (int nt = 0; nt < 512; ++nt) {
    __syncthreads();
    if (nt + 1 < 512) {                       // double-buffered prefetch
      int nb = (nt + 1) * 16;
      int j0 = threadIdx.x * 2;
      #pragma unroll
      for (int e = 0; e < 2; ++e) {
        int j = j0 + e, r = j >> 5, col = j & 31;
        smem[(nt + 1) & 1][r * 33 + col] = cb4[(size_t)(nb + r) * 32 + col];
      }
    }
    const uint4* s = smem[nt & 1];
    int ncand = nt * 16 + l16;
    float dn = cn[ncand];
    v8f acc0 = {}, acc1 = {};
    #pragma unroll
    for (int kt = 0; kt < 8; ++kt) {
      Frag bf;                                // K base + hl*8, K base + 16 + hl*8
      bf.u4[0] = s[l16 * 33 + kt * 4 + hl];
      bf.u4[1] = s[l16 * 33 + kt * 4 + 2 + hl];
      acc0 = __builtin_amdgcn_wmma_f32_16x16x32_f16(
          false, a0[kt].v, false, bf.v, (short)0, acc0, false, false);
      acc1 = __builtin_amdgcn_wmma_f32_16x16x32_f16(
          false, a1[kt].v, false, bf.v, (short)0, acc1, false, false);
    }
    #pragma unroll
    for (int v = 0; v < 8; ++v) {             // d = ||c||^2 - 2 z.c  (||z||^2 const)
      float d0 = fmaf(-2.0f, acc0[v], dn);
      bool b0 = d0 < bestD[0][v];
      bestD[0][v] = b0 ? d0 : bestD[0][v];
      bestI[0][v] = b0 ? ncand : bestI[0][v];
      float d1 = fmaf(-2.0f, acc1[v], dn);
      bool b1 = d1 < bestD[1][v];
      bestD[1][v] = b1 ? d1 : bestD[1][v];
      bestI[1][v] = b1 ? ncand : bestI[1][v];
    }
  }

  // reduce (min, argmin) across the 16 lanes holding the same row
  #pragma unroll
  for (int t = 0; t < 2; ++t) {
    #pragma unroll
    for (int v = 0; v < 8; ++v) {
      #pragma unroll
      for (int off = 1; off < 16; off <<= 1) {
        float od = __shfl_xor(bestD[t][v], off, 32);
        int   oi = __shfl_xor(bestI[t][v], off, 32);
        if (od < bestD[t][v] || (od == bestD[t][v] && oi < bestI[t][v])) {
          bestD[t][v] = od; bestI[t][v] = oi;
        }
      }
      if (l16 == 0) idx[rowBase + t * 16 + hl * 8 + v] = bestI[t][v];
    }
  }
}

// gather codebook rows back to [B,C,D,H,W] + fused loss
__global__ __launch_bounds__(256) void k_out(const float* __restrict__ z,
                                             const float* __restrict__ cb,
                                             const int* __restrict__ idx,
                                             float* __restrict__ out,
                                             float* __restrict__ loss) {
  int bc = blockIdx.x;                        // b*256 + c
  int c = bc & 255, b = bc >> 8;
  const float* zrow = z + (size_t)bc * DHW;
  float* orow = out + (size_t)bc * DHW;
  const int* idn = idx + b * DHW;
  float s = 0.0f;
  for (int i = threadIdx.x; i < DHW; i += 256) {
    int k = idn[i];
    float cv = cb[(size_t)k * C_DIM + c];     // 8MB table -> L2/WGP$ hits
    float zv = zrow[i];
    orow[i] = cv;                             // straight-through value == z_q
    float df = cv - zv;
    s += df * df;
  }
  s = blockReduceSum(s);
  if (threadIdx.x == 0)
    atomicAdd(loss, s * (2.0f / (float)OUT_ELEMS));  // (1+BETA) * mean
}

extern "C" void kernel_launch(void* const* d_in, const int* in_sizes, int n_in,
                              void* d_out, int out_size, void* d_ws, size_t ws_size,
                              hipStream_t stream) {
  (void)in_sizes; (void)n_in; (void)out_size; (void)ws_size;
  const float* z  = (const float*)d_in[0];
  const float* cb = (const float*)d_in[1];

  float* outF = (float*)d_out;
  float* loss = outF + OUT_ELEMS;
  int*   idx  = (int*)(outF + OUT_ELEMS + 1);

  char* ws = (char*)d_ws;
  _Float16* zh = (_Float16*)ws;                               // 16 MB
  _Float16* ch = (_Float16*)(ws + 16u * 1024 * 1024);         //  4 MB
  float*    cn = (float*)   (ws + 20u * 1024 * 1024);         // 32 KB

  k_cbconv<<<K_NUM, 256, 0, stream>>>(cb, ch, cn);
  dim3 gz(DHW / 64, C_DIM / 64, B_SZ);
  k_zconv<<<gz, 256, 0, stream>>>(z, zh);
  k_argmin<<<128, 256, 0, stream>>>(zh, ch, cn, idx);
  hipMemsetAsync(loss, 0, sizeof(float), stream);
  k_out<<<B_SZ * C_DIM, 256, 0, stream>>>(z, cb, idx, outF, loss);
}
